// MultiHeadAttention_87832081203628
// MI455X (gfx1250) — matrix operand order repaired and checked by reading, not compiled, not run on hardware
//
#include <hip/hip_runtime.h>
#include <hip/hip_bf16.h>

// ---------------- problem constants ----------------
constexpr int NB = 2;      // batch
constexpr int NS = 4096;   // sequence
constexpr int ND = 512;    // model dim
constexpr int NH = 8;      // heads
constexpr int NDK = 64;    // head dim (= DV)

typedef __attribute__((ext_vector_type(16))) _Float16 v16h;
typedef __attribute__((ext_vector_type(8)))  _Float16 v8h;
typedef __attribute__((ext_vector_type(8)))  float    v8f;
typedef __attribute__((ext_vector_type(4)))  unsigned int v4u;
typedef __attribute__((ext_vector_type(8)))  int          v8i;
typedef __attribute__((ext_vector_type(4)))  int          v4i;

#define DEVFN static __device__ __forceinline__

// ---------------- WMMA fragment helpers (wave32, 16x16x32 f16) ----------------
// A (16x32, row-major, ld in halves): lane holds row M=lane&15,
// K = (lane>>4)*8 + {0..7} and {16..23}.
DEVFN v16h frag_a_load(const _Float16* p, int ld) {
  int lane = threadIdx.x & 31;
  const _Float16* r = p + (lane & 15) * ld + ((lane >> 4) << 3);
  v8h x = *(const v8h*)r;
  v8h y = *(const v8h*)(r + 16);
  v16h a;
#pragma unroll
  for (int i = 0; i < 8; ++i) { a[i] = x[i]; a[i + 8] = y[i]; }
  return a;
}

// B (32x16) loaded from an N-major buffer Bt[n][k] (ld in halves):
// lane holds column N=lane&15, K = (lane>>4)*8 + {0..7} and {16..23}.
DEVFN v16h frag_b_load(const _Float16* p, int ld) {
  int lane = threadIdx.x & 31;
  const _Float16* r = p + (lane & 15) * ld + ((lane >> 4) << 3);
  v8h x = *(const v8h*)r;
  v8h y = *(const v8h*)(r + 16);
  v16h b;
#pragma unroll
  for (int i = 0; i < 8; ++i) { b[i] = x[i]; b[i + 8] = y[i]; }
  return b;
}

DEVFN v8f wmma_f16(v16h a, v16h b, v8f c) {
  return __builtin_amdgcn_wmma_f32_16x16x32_f16(false, a, false, b,
                                                (short)0, c, false, false);
}

DEVFN v8h pack8_f16(float4 a, float4 b) {
  v8h h;
  h[0] = (_Float16)a.x; h[1] = (_Float16)a.y;
  h[2] = (_Float16)a.z; h[3] = (_Float16)a.w;
  h[4] = (_Float16)b.x; h[5] = (_Float16)b.y;
  h[6] = (_Float16)b.z; h[7] = (_Float16)b.w;
  return h;
}

// ---------------- Tensor Data Mover: 2D f16 tile -> LDS (D# per ISA ch.8) ------
// Generic LDS pointer: low 32 bits are the LDS byte offset (aperture layout).
DEVFN unsigned lds_addr_of(const void* p) { return (unsigned)(size_t)p; }

// Loads a tile of tile_d1 rows x tile_d0 f16 elements from a row-major tensor
// (row stride = stride0 elements) into LDS. With pad=true, inserts 8 DWORDs
// (16 halves) of padding after every 32 DWORDs (64 halves): LDS rows become
// 80 halves (160B, 16B aligned) -> matches LDT/LDK below.
DEVFN void tdm_load_tile_f16(const _Float16* gsrc, unsigned lds_byte,
                             unsigned tensor_d0, unsigned tensor_d1,
                             unsigned tile_d0, unsigned tile_d1,
                             unsigned stride0, bool pad) {
  unsigned long long ga = (unsigned long long)(size_t)gsrc;
  v4u g0;
  g0[0] = 1u;                                   // count=1, user descriptor
  g0[1] = lds_byte;                             // LDS destination (bytes)
  g0[2] = (unsigned)ga;                         // global_addr[31:0]
  g0[3] = (unsigned)(ga >> 32) | (2u << 30);    // global_addr[56:32] | type=2
  unsigned flags = (1u << 16);                  // data_size = 2 bytes
  if (pad) flags |= (1u << 20) | (4u << 22) | (7u << 25);  // +8 dw per 32 dw
  v8i g1;
  g1[0] = (int)flags;                           // wg_mask=0 | ds=2B | pad cfg
  g1[1] = (int)(tensor_d0 << 16);               // atomic_barrier=0 | dim0 lo16
  g1[2] = (int)((tensor_d0 >> 16) | (tensor_d1 << 16));
  g1[3] = (int)((tensor_d1 >> 16) | (tile_d0 << 16));
  g1[4] = (int)(tile_d1 & 0xffffu);             // tile_dim2 = 0 (2D)
  g1[5] = (int)stride0;                         // dim0 stride lo32
  g1[6] = 0;                                    // stride hi bits = 0
  g1[7] = 0;
  v4i z4 = {};
  v8i z8 = {};
  __builtin_amdgcn_tensor_load_to_lds(g0, g1, z4, z4, z8, 0);
}

// ---------------- K0: convert + transpose weights to f16 (N-major) ------------
__global__ __launch_bounds__(256) void convert_weights_kernel(
    const float* __restrict__ Wq, const float* __restrict__ Wk,
    const float* __restrict__ Wv, const float* __restrict__ Wo,
    _Float16* __restrict__ Wtq, _Float16* __restrict__ Wtk,
    _Float16* __restrict__ Wtv, _Float16* __restrict__ Wto) {
  int idx = blockIdx.x * blockDim.x + threadIdx.x;     // 4 * 512 * 512 total
  int w  = idx >> 18;
  int rc = idx & ((1 << 18) - 1);
  int r  = rc >> 9, c = rc & 511;
  const float* src = (w == 0) ? Wq : (w == 1) ? Wk : (w == 2) ? Wv : Wo;
  _Float16*    dst = (w == 0) ? Wtq : (w == 1) ? Wtk : (w == 2) ? Wtv : Wto;
  dst[c * 512 + r] = (_Float16)src[r * 512 + c];       // transpose: Wt[n][k]
}

// ---------------- K1: QKV projection GEMM (f32 in -> f16 staged -> WMMA) ------
// C[8192 x 512] = X[8192 x 512] * W[512 x 512] + bias; per-head output layouts:
//   z=0: Qh[bh][s][dk]   z=1: Kh[bh][s][dk]   z=2: Vt[bh][dv][s] (transposed)
// B tile arrives via TDM (overlaps the VALU f32->f16 A conversion).
__global__ __launch_bounds__(128) void proj_kernel(
    const float* __restrict__ xq, const float* __restrict__ xk,
    const float* __restrict__ xv,
    const _Float16* __restrict__ wtq, const _Float16* __restrict__ wtk,
    const _Float16* __restrict__ wtv,
    const float* __restrict__ bq, const float* __restrict__ bk,
    const float* __restrict__ bv,
    _Float16* __restrict__ Qh, _Float16* __restrict__ Kh,
    _Float16* __restrict__ Vt) {
  constexpr int LDT = 80;                       // padded row, 160B = 16B aligned
  __shared__ __attribute__((aligned(16))) _Float16 As[64 * LDT];
  __shared__ __attribute__((aligned(16))) _Float16 Bs[64 * LDT];

  int z = blockIdx.z;
  const float*    X    = (z == 0) ? xq : (z == 1) ? xk : xv;
  const _Float16* Wt   = (z == 0) ? wtq : (z == 1) ? wtk : wtv;
  const float*    bias = (z == 0) ? bq : (z == 1) ? bk : bv;

  int tid = threadIdx.x, wave = tid >> 5, lane = tid & 31;
  int m0 = blockIdx.x * 64, n0 = blockIdx.y * 64;
  v8f acc[4] = {};

  for (int kc = 0; kc < ND; kc += 64) {
    if (wave == 0)                               // async B tile: Wt[n0..+64][kc..+64]
      tdm_load_tile_f16(Wt + (size_t)n0 * ND + kc, lds_addr_of(Bs),
                        ND, ND, 64, 64, ND, true);
    { // stage A tile: f32 -> f16; independent loads first, packed 16B stores
      int row = tid >> 1, c0 = (tid & 1) * 32;
      const float4* src = (const float4*)(X + (size_t)(m0 + row) * ND + kc + c0);
      float4 t0 = src[0], t1 = src[1], t2 = src[2], t3 = src[3];
      float4 t4 = src[4], t5 = src[5], t6 = src[6], t7 = src[7];
      _Float16* dst = As + row * LDT + c0;
      *(v8h*)(dst +  0) = pack8_f16(t0, t1);
      *(v8h*)(dst +  8) = pack8_f16(t2, t3);
      *(v8h*)(dst + 16) = pack8_f16(t4, t5);
      *(v8h*)(dst + 24) = pack8_f16(t6, t7);
    }
    __syncthreads();
    if (wave == 0) __builtin_amdgcn_s_wait_tensorcnt(0);
    __syncthreads();
#pragma unroll
    for (int kk = 0; kk < 64; kk += 32) {
      v16h a = frag_a_load(As + wave * 16 * LDT + kk, LDT);
#pragma unroll
      for (int nf = 0; nf < 4; ++nf) {
        v16h b = frag_b_load(Bs + nf * 16 * LDT + kk, LDT);
        acc[nf] = wmma_f16(a, b, acc[nf]);
      }
    }
    __syncthreads();
  }

  int col = lane & 15, r0 = (lane >> 4) * 8;
#pragma unroll
  for (int nf = 0; nf < 4; ++nf) {
    int n = n0 + nf * 16 + col;
    int h = n >> 6, d = n & 63;
    float bz = bias[n];
#pragma unroll
    for (int r = 0; r < 8; ++r) {
      int row = m0 + wave * 16 + r0 + r;
      int b = row >> 12, s = row & (NS - 1);
      _Float16 hv = (_Float16)(acc[nf][r] + bz);
      if (z == 0)      Qh[((size_t)(b * NH + h) * NS + s) * NDK + d] = hv;
      else if (z == 1) Kh[((size_t)(b * NH + h) * NS + s) * NDK + d] = hv;
      else             Vt[((size_t)(b * NH + h) * NDK + d) * NS + s] = hv;
    }
  }
}

// ---------------- K2: column-softmax stats (softmax over the QUERY axis) ------
// For each (b,h), key column k: m[k]=max_q s(q,k), l[k]=sum_q exp(s-m).
// One wave owns 16 key columns. Q tiles (64 rows) are TDM-staged into LDS,
// double-buffered, shared by all 8 waves.
__global__ __launch_bounds__(256) void colstats_kernel(
    const _Float16* __restrict__ Qh, const _Float16* __restrict__ Kh,
    float* __restrict__ mArr, float* __restrict__ lArr) {
  constexpr int LDQ = 80;
  __shared__ __attribute__((aligned(16))) _Float16 Qs[2][64 * LDQ];
  int bh = blockIdx.y;
  int wave = threadIdx.x >> 5, lane = threadIdx.x & 31;
  int col0 = blockIdx.x * 128 + wave * 16;
  const _Float16* Qb = Qh + (size_t)bh * NS * NDK;
  const _Float16* Kb = Kh + (size_t)bh * NS * NDK;

  // key-column B fragments are loop-invariant
  v16h bf0 = frag_b_load(Kb + (size_t)col0 * NDK + 0,  NDK);
  v16h bf1 = frag_b_load(Kb + (size_t)col0 * NDK + 32, NDK);

  if (wave == 0) {
    tdm_load_tile_f16(Qb, lds_addr_of(Qs[0]), NDK, NS, NDK, 64, NDK, true);
    __builtin_amdgcn_s_wait_tensorcnt(0);
  }
  __syncthreads();

  const float scale = 0.125f;  // 1/sqrt(64)
  float m = -3.0e38f, l = 0.f;
  int buf = 0;
  for (int q0 = 0; q0 < NS; q0 += 64) {
    if (wave == 0 && q0 + 64 < NS)
      tdm_load_tile_f16(Qb + (size_t)(q0 + 64) * NDK, lds_addr_of(Qs[buf ^ 1]),
                        NDK, NS, NDK, 64, NDK, true);
    const _Float16* qs = Qs[buf];
#pragma unroll
    for (int sub = 0; sub < 4; ++sub) {
      v16h a0 = frag_a_load(qs + (sub * 16) * LDQ + 0,  LDQ);
      v16h a1 = frag_a_load(qs + (sub * 16) * LDQ + 32, LDQ);
      v8f acc = {};
      acc = wmma_f16(a0, bf0, acc);
      acc = wmma_f16(a1, bf1, acc);
      float tmax = -3.0e38f;
#pragma unroll
      for (int r = 0; r < 8; ++r) tmax = fmaxf(tmax, acc[r] * scale);
      float mnew = fmaxf(m, tmax);
      float sum = 0.f;
#pragma unroll
      for (int r = 0; r < 8; ++r) sum += __expf(acc[r] * scale - mnew);
      l = l * __expf(m - mnew) + sum;
      m = mnew;
    }
    __syncthreads();
    if (wave == 0) __builtin_amdgcn_s_wait_tensorcnt(0);
    __syncthreads();
    buf ^= 1;
  }
  // lanes l and l+16 hold the same column (different row halves): combine
  float mo = __shfl_xor(m, 16, 32);
  float lo = __shfl_xor(l, 16, 32);
  float M = fmaxf(m, mo);
  float L = l * __expf(m - M) + lo * __expf(mo - M);
  if (lane < 16) {
    mArr[(size_t)bh * NS + col0 + lane] = M;
    lArr[(size_t)bh * NS + col0 + lane] = L;
  }
}

// ---------------- K3: out = softmax(scores) @ V with final stats --------------
// K and V tiles (64 keys) are TDM-staged into LDS, double-buffered, shared by
// all 8 waves; P tile relayout (C->A fragment) goes through per-wave LDS.
__global__ __launch_bounds__(256) void attn_out_kernel(
    const _Float16* __restrict__ Qh, const _Float16* __restrict__ Kh,
    const _Float16* __restrict__ Vt, const float* __restrict__ mArr,
    const float* __restrict__ lArr, _Float16* __restrict__ AO) {
  constexpr int LDK = 80;
  constexpr int LDP = 40;                          // 80B rows, 16B aligned
  __shared__ __attribute__((aligned(16))) _Float16 Ks[2][64 * LDK]; // [key][dk]
  __shared__ __attribute__((aligned(16))) _Float16 Vs[2][64 * LDK]; // [dv][key]
  __shared__ __attribute__((aligned(16))) _Float16 pst[8][16 * LDP];
  int bh = blockIdx.y;
  int b = bh >> 3, h = bh & 7;
  int wave = threadIdx.x >> 5, lane = threadIdx.x & 31;
  int q0 = blockIdx.x * 128 + wave * 16;
  const _Float16* Qb = Qh + (size_t)bh * NS * NDK;
  const _Float16* Kb = Kh + (size_t)bh * NS * NDK;
  const _Float16* Vb = Vt + (size_t)bh * NDK * NS;
  const float* mb = mArr + (size_t)bh * NS;
  const float* lb = lArr + (size_t)bh * NS;
  _Float16* ps = pst[wave];

  // Q fragments are loop-invariant for this wave's 16 query rows
  v16h qa0 = frag_a_load(Qb + (size_t)q0 * NDK + 0,  NDK);
  v16h qa1 = frag_a_load(Qb + (size_t)q0 * NDK + 32, NDK);
  v8f acc[4] = {};
  const float scale = 0.125f;
  int col = lane & 15, r0 = (lane >> 4) * 8;

  if (wave == 0) {
    tdm_load_tile_f16(Kb, lds_addr_of(Ks[0]), NDK, NS, NDK, 64, NDK, true);
    tdm_load_tile_f16(Vb, lds_addr_of(Vs[0]), NS, NDK, 64, NDK, NS, true);
    __builtin_amdgcn_s_wait_tensorcnt(0);
  }
  __syncthreads();

  int buf = 0;
  for (int k0 = 0; k0 < NS; k0 += 64) {
    if (wave == 0 && k0 + 64 < NS) {      // prefetch next K/V tiles via TDM
      tdm_load_tile_f16(Kb + (size_t)(k0 + 64) * NDK, lds_addr_of(Ks[buf ^ 1]),
                        NDK, NS, NDK, 64, NDK, true);
      tdm_load_tile_f16(Vb + (size_t)(k0 + 64), lds_addr_of(Vs[buf ^ 1]),
                        NS, NDK, 64, NDK, NS, true);
    }
    const _Float16* ks = Ks[buf];
    const _Float16* vs = Vs[buf];
#pragma unroll
    for (int kk = 0; kk < 64; kk += 32) {
      // two 16x16 score tiles: key cols [kk,kk+16) and [kk+16,kk+32)
      v8f s0 = {}, s1 = {};
      {
        v16h kb0 = frag_b_load(ks + (kk + 0) * LDK + 0,  LDK);
        v16h kb1 = frag_b_load(ks + (kk + 0) * LDK + 32, LDK);
        s0 = wmma_f16(qa0, kb0, s0);
        s0 = wmma_f16(qa1, kb1, s0);
        v16h kc0 = frag_b_load(ks + (kk + 16) * LDK + 0,  LDK);
        v16h kc1 = frag_b_load(ks + (kk + 16) * LDK + 32, LDK);
        s1 = wmma_f16(qa0, kc0, s1);
        s1 = wmma_f16(qa1, kc1, s1);
      }
      int kg = k0 + kk;
      float m0v = mb[kg + col],      il0 = 1.0f / lb[kg + col];
      float m1v = mb[kg + 16 + col], il1 = 1.0f / lb[kg + 16 + col];
#pragma unroll
      for (int r = 0; r < 8; ++r) {
        float p0 = __expf(s0[r] * scale - m0v) * il0;
        float p1 = __expf(s1[r] * scale - m1v) * il1;
        ps[(r0 + r) * LDP + col]      = (_Float16)p0;
        ps[(r0 + r) * LDP + 16 + col] = (_Float16)p1;
      }
      // wave-internal LDS RAW: make stores visible before cross-lane reads
      asm volatile("s_wait_dscnt 0" ::: "memory");
      v16h pa = frag_a_load(ps, LDP);
#pragma unroll
      for (int nf = 0; nf < 4; ++nf) {
        v16h vb = frag_b_load(vs + (nf * 16) * LDK + kk, LDK);
        acc[nf] = wmma_f16(pa, vb, acc[nf]);
      }
    }
    __syncthreads();
    if (wave == 0) __builtin_amdgcn_s_wait_tensorcnt(0);
    __syncthreads();
    buf ^= 1;
  }
  // write attn output token-major [b, s, h*64+d] as f16 for the final GEMM
#pragma unroll
  for (int nf = 0; nf < 4; ++nf) {
    int d = nf * 16 + col;
#pragma unroll
    for (int r = 0; r < 8; ++r) {
      int q = q0 + r0 + r;
      AO[((size_t)(b * NS + q)) * ND + h * NDK + d] = (_Float16)acc[nf][r];
    }
  }
}

// ---------------- K4: output projection GEMM (f16 x f16 -> f32 out) -----------
// Both A and B tiles arrive via TDM, double-buffered.
__global__ __launch_bounds__(128) void out_proj_kernel(
    const _Float16* __restrict__ AO, const _Float16* __restrict__ Wot,
    const float* __restrict__ bo, float* __restrict__ Y) {
  constexpr int LDT = 80;
  __shared__ __attribute__((aligned(16))) _Float16 As[2][64 * LDT];
  __shared__ __attribute__((aligned(16))) _Float16 Bs[2][64 * LDT];
  int tid = threadIdx.x, wave = tid >> 5, lane = tid & 31;
  int m0 = blockIdx.x * 64, n0 = blockIdx.y * 64;
  v8f acc[4] = {};

  if (wave == 0) {
    tdm_load_tile_f16(AO + (size_t)m0 * ND, lds_addr_of(As[0]),
                      ND, NB * NS, 64, 64, ND, true);
    tdm_load_tile_f16(Wot + (size_t)n0 * ND, lds_addr_of(Bs[0]),
                      ND, ND, 64, 64, ND, true);
    __builtin_amdgcn_s_wait_tensorcnt(0);
  }
  __syncthreads();

  int buf = 0;
  for (int kc = 0; kc < ND; kc += 64) {
    if (wave == 0 && kc + 64 < ND) {
      tdm_load_tile_f16(AO + (size_t)m0 * ND + kc + 64, lds_addr_of(As[buf ^ 1]),
                        ND, NB * NS, 64, 64, ND, true);
      tdm_load_tile_f16(Wot + (size_t)n0 * ND + kc + 64, lds_addr_of(Bs[buf ^ 1]),
                        ND, ND, 64, 64, ND, true);
    }
#pragma unroll
    for (int kk = 0; kk < 64; kk += 32) {
      v16h a = frag_a_load(As[buf] + wave * 16 * LDT + kk, LDT);
#pragma unroll
      for (int nf = 0; nf < 4; ++nf) {
        v16h b = frag_b_load(Bs[buf] + nf * 16 * LDT + kk, LDT);
        acc[nf] = wmma_f16(a, b, acc[nf]);
      }
    }
    __syncthreads();
    if (wave == 0) __builtin_amdgcn_s_wait_tensorcnt(0);
    __syncthreads();
    buf ^= 1;
  }

  int col = lane & 15, r0 = (lane >> 4) * 8;
#pragma unroll
  for (int nf = 0; nf < 4; ++nf) {
    int n = n0 + nf * 16 + col;
    float bz = bo[n];
#pragma unroll
    for (int r = 0; r < 8; ++r) {
      int row = m0 + wave * 16 + r0 + r;
      Y[(size_t)row * ND + n] = acc[nf][r] + bz;
    }
  }
}

// ---------------- host launch ----------------
extern "C" void kernel_launch(void* const* d_in, const int* in_sizes, int n_in,
                              void* d_out, int out_size, void* d_ws, size_t ws_size,
                              hipStream_t stream) {
  (void)in_sizes; (void)n_in; (void)out_size; (void)ws_size;
  const float* q  = (const float*)d_in[0];
  const float* k  = (const float*)d_in[1];
  const float* v  = (const float*)d_in[2];
  const float* Wq = (const float*)d_in[3];
  const float* bq = (const float*)d_in[4];
  const float* Wk = (const float*)d_in[5];
  const float* bk = (const float*)d_in[6];
  const float* Wv = (const float*)d_in[7];
  const float* bv = (const float*)d_in[8];
  const float* Wo = (const float*)d_in[9];
  const float* bo = (const float*)d_in[10];
  float* out = (float*)d_out;

  char* ws = (char*)d_ws;
  size_t off = 0;
  auto wsalloc = [&](size_t bytes) -> char* {
    char* p = ws + off;
    off += (bytes + 255) & ~(size_t)255;
    return p;
  };
  _Float16* Wtq = (_Float16*)wsalloc((size_t)ND * ND * 2);
  _Float16* Wtk = (_Float16*)wsalloc((size_t)ND * ND * 2);
  _Float16* Wtv = (_Float16*)wsalloc((size_t)ND * ND * 2);
  _Float16* Wto = (_Float16*)wsalloc((size_t)ND * ND * 2);
  _Float16* Qh  = (_Float16*)wsalloc((size_t)NB * NH * NS * NDK * 2);
  _Float16* Kh  = (_Float16*)wsalloc((size_t)NB * NH * NS * NDK * 2);
  _Float16* Vt  = (_Float16*)wsalloc((size_t)NB * NH * NS * NDK * 2);
  float*    mA  = (float*)wsalloc((size_t)NB * NH * NS * 4);
  float*    lA  = (float*)wsalloc((size_t)NB * NH * NS * 4);
  _Float16* AO  = (_Float16*)wsalloc((size_t)NB * NS * ND * 2);

  convert_weights_kernel<<<(4 * ND * ND) / 256, 256, 0, stream>>>(
      Wq, Wk, Wv, Wo, Wtq, Wtk, Wtv, Wto);

  proj_kernel<<<dim3((NB * NS) / 64, ND / 64, 3), 128, 0, stream>>>(
      q, k, v, Wtq, Wtk, Wtv, bq, bk, bv, Qh, Kh, Vt);

  colstats_kernel<<<dim3(NS / 128, NB * NH), 256, 0, stream>>>(Qh, Kh, mA, lA);

  attn_out_kernel<<<dim3(NS / 128, NB * NH), 256, 0, stream>>>(
      Qh, Kh, Vt, mA, lA, AO);

  out_proj_kernel<<<dim3((NB * NS) / 64, ND / 64), 128, 0, stream>>>(
      AO, Wto, bo, out);
}
